// CrossInferenceBlock_5677946765982
// MI455X (gfx1250) — matrix operand order repaired and checked by reading, not compiled, not run on hardware
//
#include <hip/hip_runtime.h>

#define T_DIM 256
#define S_DIM 256
#define F_DIM 1024
#define A_DIM 256

typedef __attribute__((ext_vector_type(16))) __bf16 v16bf;
typedef __attribute__((ext_vector_type(8)))  __bf16 v8bf;
typedef __attribute__((ext_vector_type(8)))  float  v8f;

__device__ __forceinline__ __bf16 f2bf(float f) { return (__bf16)f; }  // hw v_cvt

__device__ __forceinline__ void st_out(float* p, float v)  { *p = v; }
__device__ __forceinline__ void st_out(__bf16* p, float v) { *p = f2bf(v); }

// ---------------------------------------------------------------------------
// Kernel 0: W (K x N, fp32) -> W^T (N x K, bf16)
// ---------------------------------------------------------------------------
__global__ __launch_bounds__(256) void wt_transpose_kernel(
    const float* __restrict__ w, __bf16* __restrict__ wt, int K, int N) {
  int idx = blockIdx.x * 256 + threadIdx.x;       // idx = n*K + k
  int n = idx / K;
  int k = idx - n * K;
  if (n < N) wt[idx] = f2bf(w[(size_t)k * N + n]);
}

// ---------------------------------------------------------------------------
// Unified pipelined WMMA GEMM:
//   C_b = A_b(M x K, AT) * Bt_b(N x K bf16)^T * scale (+ bias)
// A,Bt row-major, ld == K. Batched via blockIdx.z (strides sA/sB/sC).
// Block tile 128x128, BK=64 (2 WMMA K-substeps), 8 waves x (32x64 sub-tile),
// double-buffered LDS, one barrier per K-iteration.
// TOUT: store transposed per t-chunk: out[(t*N + n)*S + s]  (bf16 only).
// ---------------------------------------------------------------------------
template<typename AT, typename OT, bool TOUT, bool HAS_BIAS>
__global__ __launch_bounds__(256) void gemm_kernel(
    const AT*     __restrict__ Ag,
    const __bf16* __restrict__ Bg,
    const float*  __restrict__ bias,
    OT*           __restrict__ Cg,
    int N, int K,
    long long sA, long long sB, long long sC,
    float scale) {
  constexpr int BM = 128, BN = 128, BK = 64, LDT = BK + 8;  // 144B LDS rows
  constexpr int TILE = BM * LDT;
  constexpr bool A_F32 = (sizeof(AT) == 4);
  __shared__ __bf16 As[2 * TILE];
  __shared__ __bf16 Bs[2 * TILE];

  const AT*     A  = Ag + (size_t)blockIdx.z * sA;
  const __bf16* Bt = Bg + (size_t)blockIdx.z * sB;
  OT*           C  = Cg + (size_t)blockIdx.z * sC;

  const int tid   = threadIdx.x;
  const int lane  = tid & 31;
  const int wid   = tid >> 5;
  const int wm    = wid >> 1;      // 0..3  M sub-tile
  const int wn    = wid & 1;       // 0..1  N sub-tile
  const int lrow  = lane & 15;
  const int lhalf = lane >> 4;
  const int m0 = blockIdx.y * BM;
  const int n0 = blockIdx.x * BN;

  const int ldr  = tid >> 1;         // 0..127 staged row
  const int lseg = (tid & 1) << 5;   // 0 or 32

  const AT*     arow = A  + (size_t)(m0 + ldr) * K + lseg;
  const __bf16* brow = Bt + (size_t)(n0 + ldr) * K + lseg;
  __bf16* asl = As + ldr * LDT + lseg;
  __bf16* bsl = Bs + ldr * LDT + lseg;

  v8f acc[2][4];
#pragma unroll
  for (int i = 0; i < 2; ++i)
#pragma unroll
    for (int j = 0; j < 4; ++j)
#pragma unroll
      for (int e = 0; e < 8; ++e) acc[i][j][e] = 0.0f;

  const int NT = K / BK;

  // staging registers
  float fa[32];     // used when A is fp32
  v8bf  ra[4];      // used when A is bf16
  v8bf  rb[4];

  // ---- prologue: fetch tile 0 ----
  if (A_F32) {
#pragma unroll
    for (int q = 0; q < 8; ++q)
      *(float4*)(fa + 4 * q) = *(const float4*)((const float*)arow + 4 * q);
  } else {
#pragma unroll
    for (int q = 0; q < 4; ++q) ra[q] = *(const v8bf*)((const __bf16*)arow + 8 * q);
  }
#pragma unroll
  for (int q = 0; q < 4; ++q) rb[q] = *(const v8bf*)(brow + 8 * q);

  int buf = 0;
  for (int kt = 0; kt < NT; ++kt) {
    // ---- commit staged tile kt to LDS[buf] ----
    __bf16* ad = asl + buf * TILE;
    __bf16* bd = bsl + buf * TILE;
    if (A_F32) {
#pragma unroll
      for (int q = 0; q < 4; ++q) {
        v8bf v;
#pragma unroll
        for (int e = 0; e < 8; ++e) v[e] = f2bf(fa[8 * q + e]);
        *(v8bf*)(ad + 8 * q) = v;
      }
    } else {
#pragma unroll
      for (int q = 0; q < 4; ++q) *(v8bf*)(ad + 8 * q) = ra[q];
    }
#pragma unroll
    for (int q = 0; q < 4; ++q) *(v8bf*)(bd + 8 * q) = rb[q];

    __syncthreads();

    // ---- issue global fetch of tile kt+1 (overlaps with WMMA below) ----
    if (kt + 1 < NT) {
      const int ko = (kt + 1) * BK;
      if (A_F32) {
#pragma unroll
        for (int q = 0; q < 8; ++q)
          *(float4*)(fa + 4 * q) = *(const float4*)((const float*)arow + ko + 4 * q);
      } else {
#pragma unroll
        for (int q = 0; q < 4; ++q)
          ra[q] = *(const v8bf*)((const __bf16*)arow + ko + 8 * q);
      }
#pragma unroll
      for (int q = 0; q < 4; ++q) rb[q] = *(const v8bf*)(brow + ko + 8 * q);
      if (kt + 2 < NT) {
        __builtin_prefetch((const char*)arow + (size_t)(kt + 2) * BK * sizeof(AT), 0, 0);
        __builtin_prefetch(brow + (kt + 2) * BK, 0, 0);
      }
    }

    // ---- compute from LDS[buf]: 2 K-substeps x 8 WMMAs ----
    const __bf16* ab = As + buf * TILE;
    const __bf16* bb = Bs + buf * TILE;
#pragma unroll
    for (int sub = 0; sub < 2; ++sub) {
      v16bf afr[2], bfr[4];
#pragma unroll
      for (int mt = 0; mt < 2; ++mt) {
        const __bf16* p = ab + (wm * 32 + mt * 16 + lrow) * LDT + sub * 32;
        union { v16bf v; v8bf h[2]; } u;
        u.h[0] = *(const v8bf*)(p + lhalf * 8);        // K = 8*half .. +7
        u.h[1] = *(const v8bf*)(p + 16 + lhalf * 8);   // K = 16+8*half .. +7
        afr[mt] = u.v;
      }
#pragma unroll
      for (int nt = 0; nt < 4; ++nt) {
        const __bf16* p = bb + (wn * 64 + nt * 16 + lrow) * LDT + sub * 32;
        union { v16bf v; v8bf h[2]; } u;
        u.h[0] = *(const v8bf*)(p + lhalf * 8);        // K = 8*half .. +7
        u.h[1] = *(const v8bf*)(p + 16 + lhalf * 8);
        bfr[nt] = u.v;
      }
#pragma unroll
      for (int mt = 0; mt < 2; ++mt)
#pragma unroll
        for (int nt = 0; nt < 4; ++nt)
          acc[mt][nt] = __builtin_amdgcn_wmma_f32_16x16x32_bf16(
              false, afr[mt], false, bfr[nt], (short)0, acc[mt][nt], false, false);
    }
    buf ^= 1;
  }

  // ---- epilogue ----
#pragma unroll
  for (int mt = 0; mt < 2; ++mt) {
    const int mbase = m0 + wm * 32 + mt * 16 + lhalf * 8;
#pragma unroll
    for (int nt = 0; nt < 4; ++nt) {
      const int col = n0 + wn * 64 + nt * 16 + lrow;
      const float bv = HAS_BIAS ? bias[col] : 0.0f;
      if (TOUT) {  // transposed per-t store: 8 contiguous bf16 per lane
        const int tt = mbase >> 8;
        const int ss = mbase & (S_DIM - 1);
        OT* op = Cg + ((size_t)tt * N + col) * S_DIM + ss;
        v8bf vv;
#pragma unroll
        for (int i = 0; i < 8; ++i) vv[i] = f2bf(acc[mt][nt][i] * scale + bv);
        *(v8bf*)op = vv;
      } else {
#pragma unroll
        for (int i = 0; i < 8; ++i)
          st_out(&C[(size_t)(mbase + i) * N + col], acc[mt][nt][i] * scale + bv);
      }
    }
  }
}

// ---------------------------------------------------------------------------
extern "C" void kernel_launch(void* const* d_in, const int* in_sizes, int n_in,
                              void* d_out, int out_size, void* d_ws, size_t ws_size,
                              hipStream_t stream) {
  (void)in_sizes; (void)n_in; (void)out_size; (void)ws_size;
  const float* batch = (const float*)d_in[0];
  const float* a_w   = (const float*)d_in[1];
  const float* a_b   = (const float*)d_in[2];
  const float* b_w   = (const float*)d_in[3];
  const float* b_b   = (const float*)d_in[4];
  const float* g_w   = (const float*)d_in[5];
  const float* g_b   = (const float*)d_in[6];
  float* out = (float*)d_out;

  char* ws = (char*)d_ws;
  size_t off = 0;
  __bf16* AwT    = (__bf16*)(ws + off); off += (size_t)A_DIM * F_DIM * 2;          // 0.5 MB
  __bf16* BwT    = (__bf16*)(ws + off); off += (size_t)A_DIM * F_DIM * 2;          // 0.5 MB
  __bf16* GwT    = (__bf16*)(ws + off); off += (size_t)F_DIM * F_DIM * 2;          // 2 MB
  __bf16* theta  = (__bf16*)(ws + off); off += (size_t)T_DIM * S_DIM * A_DIM * 2;  // 32 MB
  __bf16* phi    = (__bf16*)(ws + off); off += (size_t)T_DIM * S_DIM * A_DIM * 2;  // 32 MB
  __bf16* attn   = (__bf16*)(ws + off); off += (size_t)T_DIM * S_DIM * S_DIM * 2;  // 32 MB
  __bf16* featsT = (__bf16*)(ws + off); off += (size_t)T_DIM * F_DIM * S_DIM * 2;  // 128 MB

  const int M = T_DIM * S_DIM;  // 65536

  // Stage 0: weight transpose + bf16 convert
  wt_transpose_kernel<<<(A_DIM * F_DIM) / 256, 256, 0, stream>>>(a_w, AwT, F_DIM, A_DIM);
  wt_transpose_kernel<<<(A_DIM * F_DIM) / 256, 256, 0, stream>>>(b_w, BwT, F_DIM, A_DIM);
  wt_transpose_kernel<<<(F_DIM * F_DIM) / 256, 256, 0, stream>>>(g_w, GwT, F_DIM, F_DIM);

  // Stage 1: projections (A fp32 converted in-kernel, W^T bf16)
  gemm_kernel<float, __bf16, false, true><<<dim3(A_DIM / 128, M / 128, 1), 256, 0, stream>>>(
      batch, AwT, a_b, theta, A_DIM, F_DIM, 0, 0, 0, 1.0f);
  gemm_kernel<float, __bf16, false, true><<<dim3(A_DIM / 128, M / 128, 1), 256, 0, stream>>>(
      batch, BwT, b_b, phi, A_DIM, F_DIM, 0, 0, 0, 1.0f);
  gemm_kernel<float, __bf16, true, true><<<dim3(F_DIM / 128, M / 128, 1), 256, 0, stream>>>(
      batch, GwT, g_b, featsT, F_DIM, F_DIM, 0, 0, 0, 1.0f);

  // Stage 2: attn_t = theta_t @ phi_t^T, 1/(s+t) folded here
  gemm_kernel<__bf16, __bf16, false, false><<<dim3(S_DIM / 128, S_DIM / 128, T_DIM), 256, 0, stream>>>(
      theta, phi, nullptr, attn, S_DIM, A_DIM,
      (long long)S_DIM * A_DIM, (long long)S_DIM * A_DIM, (long long)S_DIM * S_DIM,
      1.0f / (float)(S_DIM + T_DIM));

  // Stage 3: out_t = attn_t @ feats_t (B operand = featsT_t: F x S, contiguous-K)
  gemm_kernel<__bf16, float, false, false><<<dim3(F_DIM / 128, S_DIM / 128, T_DIM), 256, 0, stream>>>(
      attn, featsT, nullptr, out, F_DIM, S_DIM,
      (long long)S_DIM * S_DIM, (long long)F_DIM * S_DIM, (long long)S_DIM * F_DIM,
      1.0f);
}
